// RPEMetaLA_39986145525843
// MI455X (gfx1250) — matrix operand order repaired and checked by reading, not compiled, not run on hardware
//
#include <hip/hip_runtime.h>
#include <hip/hip_bf16.h>
#include <math.h>

// ---------------------------------------------------------------- constants
#define BB      2
#define LL      512
#define EMBED   512
#define DMODEL  256
#define HH      8
#define DKDIM   256      // q / k_gate dim
#define HEADV   64       // head_dim (v)
#define KEYD    32       // key_dim
#define DCONV   4
#define NPROJ   1536     // q(256) + kg(256) + v(512) + g(512)
#define MROWS   (BB*LL)  // 1024
#define CHUNK   64       // GLA chunk length
#define SCALEQ  0.17677669529663687f  // 32^-0.5

typedef __attribute__((ext_vector_type(16))) _Float16 v16h;
typedef __attribute__((ext_vector_type(8)))  _Float16 v8h;
typedef __attribute__((ext_vector_type(8)))  float    v8f;

// ----------------------------------------------------------- WMMA fragments
// A-fragment (16-bit A 16x32 ISA layout): rowptr = &A[m0+lo][k0]
//   lane<16: elems K = k0+0..7, k0+16..23 ; lane>=16: +8 shift
__device__ __forceinline__ v16h afrag(const _Float16* rowptr, int kh2) {
    v8h a0 = *(const v8h*)(rowptr + 8 * kh2);
    v8h a1 = *(const v8h*)(rowptr + 8 * kh2 + 16);
    v16h r;
#pragma unroll
    for (int i = 0; i < 8; ++i) { r[i] = a0[i]; r[i + 8] = a1[i]; }
    return r;
}
// B-fragment: rowptr = &Bw[n0+lo][k0] (n-major storage, contiguous K)
//   lane<16: K = k0+0..7, k0+16..23 ; lane>=16: +8 shift
__device__ __forceinline__ v16h bfrag(const _Float16* rowptr, int kh2) {
    v8h b0 = *(const v8h*)(rowptr + 8 * kh2);
    v8h b1 = *(const v8h*)(rowptr + 8 * kh2 + 16);
    v16h r;
#pragma unroll
    for (int i = 0; i < 8; ++i) { r[i] = b0[i]; r[i + 8] = b1[i]; }
    return r;
}
#define WMMA(a, b, c) __builtin_amdgcn_wmma_f32_16x16x32_f16(false, (a), false, (b), (short)0, (c), false, false)

// ---------------------------------------------------- 1. causal dwconv + SiLU
__global__ void conv_silu_kernel(const float* __restrict__ x,
                                 const float* __restrict__ w,
                                 _Float16* __restrict__ xs_h) {
    int idx = blockIdx.x * blockDim.x + threadIdx.x;
    if (idx >= BB * LL * EMBED) return;
    int d = idx & (EMBED - 1);
    int l = (idx / EMBED) & (LL - 1);
    int b = idx / (EMBED * LL);
    float acc = 0.f;
#pragma unroll
    for (int j = 0; j < DCONV; ++j) {
        int ls = l + j - (DCONV - 1);
        if (ls >= 0) acc += x[((size_t)(b * LL + ls)) * EMBED + d] * w[d * DCONV + j];
    }
    float s = acc / (1.f + __expf(-acc));
    xs_h[idx] = (_Float16)s;
}

// ------------------------------------- 2. mean over l2 of embed_qk (536 MB!)
__global__ void reduce_embed_kernel(const float* __restrict__ e,
                                    _Float16* __restrict__ ered_h) {
    int bl = blockIdx.x;                                      // 0..1023
    const float* base = e + (size_t)bl * LL * DMODEL;
    int r  = threadIdx.x >> 6;
    int c4 = threadIdx.x & 63;
    float4 acc = make_float4(0.f, 0.f, 0.f, 0.f);
    for (int l2 = r; l2 < LL; l2 += 4) {
        const float* p = base + (size_t)l2 * DMODEL + c4 * 4;
        __builtin_prefetch(p + 4 * DMODEL, 0, 0);
        float4 v = *(const float4*)p;
        acc.x += v.x; acc.y += v.y; acc.z += v.z; acc.w += v.w;
    }
    __shared__ float red[4][DMODEL];
    red[r][c4 * 4 + 0] = acc.x;
    red[r][c4 * 4 + 1] = acc.y;
    red[r][c4 * 4 + 2] = acc.z;
    red[r][c4 * 4 + 3] = acc.w;
    __syncthreads();
    int c = threadIdx.x;
    float s = red[0][c] + red[1][c] + red[2][c] + red[3][c];
    ered_h[(size_t)bl * DMODEL + c] = (_Float16)(s * (1.f / (float)LL));
}

// ----------------------------------------------------- 3. fp32 -> fp16 copy
__global__ void f32_to_f16_kernel(const float* __restrict__ src,
                                  _Float16* __restrict__ dst, int n) {
    int i = blockIdx.x * blockDim.x + threadIdx.x;
    if (i < n) dst[i] = (_Float16)src[i];
}

// --------------------------------------------------------- 4. WMMA f16 GEMM
// C[MxN] = A[MxK] @ B^T, B stored n-major Bw[n][k]. One wave per 16x16 tile.
__global__ void wmma_gemm_kernel(const _Float16* __restrict__ A,
                                 const _Float16* __restrict__ Bw,
                                 float* __restrict__ C,
                                 int M, int N, int K) {
    const int n0  = blockIdx.x * 16;
    const int m0  = blockIdx.y * 16;
    const int lo  = threadIdx.x & 15;
    const int kh2 = threadIdx.x >> 4;
    v8f acc = {};
    const _Float16* arow = A  + (size_t)(m0 + lo) * K;
    const _Float16* brow = Bw + (size_t)(n0 + lo) * K;
    for (int k0 = 0; k0 < K; k0 += 32)
        acc = WMMA(afrag(arow + k0, kh2), bfrag(brow + k0, kh2), acc);
    float* crow = C + (size_t)(m0 + 8 * kh2) * N + n0 + lo;
#pragma unroll
    for (int r = 0; r < 8; ++r) crow[(size_t)r * N] = acc[r];
}

// --------------------------- 5. gk = log_sigmoid(kg)/16 ; kh = 1-exp(gk)+p_red
__global__ void gate_decay_kernel(const float* __restrict__ proj,
                                  const float* __restrict__ p_red,
                                  float* __restrict__ gkbuf,
                                  float* __restrict__ khbuf) {
    int i = blockIdx.x * blockDim.x + threadIdx.x;
    if (i >= MROWS * DKDIM) return;
    int m = i >> 8;
    int d = i & 255;
    float xg = proj[(size_t)m * NPROJ + DKDIM + d];
    float ls = (xg >= 0.f) ? -log1pf(__expf(-xg)) : xg - log1pf(__expf(xg));
    float gk = ls * (1.f / 16.f);
    gkbuf[i] = gk;
    khbuf[i] = (1.f - __expf(gk)) + p_red[i];
}

// --------------------------------------- 6. chunk-parallel GLA (WMMA-based)
// One block per (b,h); 4 waves; 8 sequential chunks of 64.
// Per chunk:  A = Q~ K^^T (causal)      [Q~ = q*exp(c)*scale, K^ = k*exp(-c)]
//             O = Q~ @ S_prev + A @ V
//             S = diag(exp(cL)) S + Kbar^T V   [Kbar = k*exp(cL - c)]
__global__ void __launch_bounds__(128)
gla_chunk_kernel(const float* __restrict__ proj,
                 const float* __restrict__ gkbuf,
                 const float* __restrict__ khbuf,
                 float* __restrict__ o_scan) {
    const int bh   = blockIdx.x;              // 0..15
    const int b    = bh >> 3;
    const int h    = bh & 7;
    const int tid  = threadIdx.x;             // 0..127
    const int wave = tid >> 5;
    const int lane = tid & 31;
    const int lo   = lane & 15;
    const int kh2  = lane >> 4;

    __shared__ alignas(16) float    cbuf[CHUNK][KEYD];   // inclusive cumsum(gk)
    __shared__ alignas(16) _Float16 Qh [CHUNK][KEYD];    // q~  (t-major)
    __shared__ alignas(16) _Float16 Kxh[CHUNK][KEYD];    // k^  (s-major)
    __shared__ alignas(16) _Float16 Kt [KEYD][CHUNK];    // kbar^T (k-major)
    __shared__ alignas(16) _Float16 Vt [HEADV][CHUNK];   // V^T (v-major)
    __shared__ alignas(16) _Float16 Ah [CHUNK][CHUNK];   // masked attention
    __shared__ alignas(16) float    Sf [KEYD][HEADV];    // state fp32
    __shared__ alignas(16) _Float16 St [HEADV][KEYD];    // state^T fp16
    __shared__ float eL[KEYD];

    for (int i = tid; i < KEYD * HEADV; i += 128) {
        ((float*)Sf)[i]    = 0.f;
        ((_Float16*)St)[i] = (_Float16)0.f;
    }
    __syncthreads();

    for (int c0 = 0; c0 < LL; c0 += CHUNK) {
        const size_t mb = (size_t)b * LL + c0;

        // phase A: per-k inclusive cumsum of gk over t (coalesced across k)
        if (tid < KEYD) {
            float c = 0.f;
            for (int t = 0; t < CHUNK; ++t) {
                c += gkbuf[(mb + t) * DKDIM + h * KEYD + tid];
                cbuf[t][tid] = c;
            }
            eL[tid] = __expf(c);
        }
        __syncthreads();

        // phase B: decay-folded operands (fp16, fragment-friendly layouts)
        for (int i = tid; i < CHUNK * KEYD; i += 128) {
            int t = i >> 5, k = i & 31;
            float cc = cbuf[t][k];
            float cl = cbuf[CHUNK - 1][k];
            float qv = proj [(mb + t) * NPROJ + h * KEYD + k];
            float kv = khbuf[(mb + t) * DKDIM + h * KEYD + k];
            Qh [t][k] = (_Float16)(qv * __expf(cc) * SCALEQ);
            Kxh[t][k] = (_Float16)(kv * __expf(-cc));
            Kt [k][t] = (_Float16)(kv * __expf(cl - cc));
        }
        for (int i = tid; i < CHUNK * HEADV; i += 128) {
            int t = i >> 6, v = i & 63;
            Vt[v][t] = (_Float16)proj[(mb + t) * NPROJ + 512 + h * HEADV + v];
        }
        __syncthreads();

        // matmul 1: A = Qh @ Kxh^T, causal mask, -> Ah (fp16)
        {
            const int m0 = wave * 16;
            v16h aq = afrag(&Qh[m0 + lo][0], kh2);
            for (int nt = 0; nt < 4; ++nt) {
                const int n0 = nt * 16;
                if (n0 > m0) {                       // strictly-upper tile: zero
#pragma unroll
                    for (int r = 0; r < 8; ++r)
                        Ah[m0 + 8 * kh2 + r][n0 + lo] = (_Float16)0.f;
                    continue;
                }
                v8f acc = {};
                acc = WMMA(aq, bfrag(&Kxh[n0 + lo][0], kh2), acc);
#pragma unroll
                for (int r = 0; r < 8; ++r) {
                    float val = acc[r];
                    if (n0 == m0 && lo > 8 * kh2 + r) val = 0.f;   // s > t
                    Ah[m0 + 8 * kh2 + r][n0 + lo] = (_Float16)val;
                }
            }
        }
        __syncthreads();

        // matmul 2+3: O = Qh @ S_prev + Ah @ V  -> o_scan
        {
            const int m0 = wave * 16;
            v16h aq  = afrag(&Qh[m0 + lo][0],  kh2);
            v16h aa0 = afrag(&Ah[m0 + lo][0],  kh2);
            v16h aa1 = afrag(&Ah[m0 + lo][32], kh2);
            for (int nt = 0; nt < 4; ++nt) {
                const int n0 = nt * 16;
                v8f acc = {};
                acc = WMMA(aq,  bfrag(&St[n0 + lo][0],  kh2), acc);   // inter
                acc = WMMA(aa0, bfrag(&Vt[n0 + lo][0],  kh2), acc);   // intra
                acc = WMMA(aa1, bfrag(&Vt[n0 + lo][32], kh2), acc);
                float* orow = o_scan + ((size_t)bh * LL + c0 + m0 + 8 * kh2) * HEADV + n0 + lo;
#pragma unroll
                for (int r = 0; r < 8; ++r) orow[(size_t)r * HEADV] = acc[r];
            }
        }
        __syncthreads();                      // protect St before update

        // matmul 4: S = diag(eL) S + Kt @ V   (32x64 += 32x64x64)
        {
#pragma unroll
            for (int u = 0; u < 2; ++u) {
                int tt = wave * 2 + u;        // 8 tiles over 4 waves
                int m0 = (tt >> 2) * 16;      // k-dim tile
                int n0 = (tt & 3) * 16;       // v tile
                v8f acc = {};
                acc = WMMA(afrag(&Kt[m0 + lo][0],  kh2), bfrag(&Vt[n0 + lo][0],  kh2), acc);
                acc = WMMA(afrag(&Kt[m0 + lo][32], kh2), bfrag(&Vt[n0 + lo][32], kh2), acc);
#pragma unroll
                for (int r = 0; r < 8; ++r) {
                    int kk = m0 + 8 * kh2 + r, vv = n0 + lo;
                    float s = eL[kk] * Sf[kk][vv] + acc[r];
                    Sf[kk][vv] = s;
                    St[vv][kk] = (_Float16)s;
                }
            }
        }
        __syncthreads();
    }
}

// ---------------- 7. aug-sigmoid + LayerNorm(64) + silu(g) gate -> z (fp16)
__global__ void aug_ln_gate_kernel(const float* __restrict__ proj,
                                   const float* __restrict__ khbuf,
                                   const float* __restrict__ o_scan,
                                   const float* __restrict__ aug,
                                   const float* __restrict__ bg,
                                   _Float16* __restrict__ z_h) {
    const int idx = blockIdx.x;               // 0..8191
    const int b = idx / (HH * LL);
    const int h = (idx / LL) & 7;
    const int l = idx & (LL - 1);
    const int tid = threadIdx.x;              // 0..63
    const size_t m = (size_t)(b * LL + l);

    __shared__ float s_aug;
    __shared__ float r1[HEADV], r2[HEADV];

    float p = 0.f;
    if (tid < 32) {
        float qv = proj [m * NPROJ + h * KEYD + tid];
        float kv = khbuf[m * DKDIM + h * KEYD + tid];
        p = qv * aug[h * KEYD + tid] * kv;
#pragma unroll
        for (int off = 16; off; off >>= 1) p += __shfl_xor(p, off, 32);
        if (tid == 0) s_aug = p;
    }
    __syncthreads();

    float o  = o_scan[((size_t)(b * HH + h) * LL + l) * HEADV + tid];
    float vt = proj[m * NPROJ + 512 + h * HEADV + tid];
    float o2 = o + 1.f / (1.f + __expf(-s_aug * vt));

    r1[tid] = o2;
    r2[tid] = o2 * o2;
    __syncthreads();
#pragma unroll
    for (int off = 32; off; off >>= 1) {
        if (tid < off) { r1[tid] += r1[tid + off]; r2[tid] += r2[tid + off]; }
        __syncthreads();
    }
    float mu  = r1[0] * (1.f / 64.f);
    float var = r2[0] * (1.f / 64.f) - mu * mu;
    float on  = (o2 - mu) * rsqrtf(var + 1e-5f);

    float gv = proj[m * NPROJ + 1024 + h * HEADV + tid] + bg[h * HEADV + tid];
    float zz = gv / (1.f + __expf(-gv)) * on;
    z_h[m * EMBED + h * HEADV + tid] = (_Float16)zz;
}

// ---------------------------------------------------------------- launcher
extern "C" void kernel_launch(void* const* d_in, const int* in_sizes, int n_in,
                              void* d_out, int out_size, void* d_ws, size_t ws_size,
                              hipStream_t stream) {
    (void)in_sizes; (void)n_in; (void)out_size; (void)ws_size;
    const float* x       = (const float*)d_in[0];
    const float* embedqk = (const float*)d_in[1];
    const float* conv_w  = (const float*)d_in[2];
    const float* Wq      = (const float*)d_in[3];
    const float* Wkg     = (const float*)d_in[4];
    const float* Wv      = (const float*)d_in[5];
    const float* Wp      = (const float*)d_in[6];
    const float* Wg      = (const float*)d_in[7];
    const float* bg      = (const float*)d_in[8];
    const float* Wout    = (const float*)d_in[9];
    const float* aug     = (const float*)d_in[10];
    float* out = (float*)d_out;

    size_t off = 0;
    char* base = (char*)d_ws;
    auto carve = [&](size_t bytes) -> void* {
        void* p = base + off;
        off += (bytes + 255) & ~(size_t)255;
        return p;
    };
    _Float16* xs_h   = (_Float16*)carve((size_t)MROWS * EMBED * 2);
    _Float16* Wcat_h = (_Float16*)carve((size_t)NPROJ * EMBED * 2);
    _Float16* Wp_h   = (_Float16*)carve((size_t)DMODEL * DMODEL * 2);
    _Float16* Wout_h = (_Float16*)carve((size_t)EMBED * EMBED * 2);
    _Float16* ered_h = (_Float16*)carve((size_t)MROWS * DMODEL * 2);
    _Float16* z_h    = (_Float16*)carve((size_t)MROWS * EMBED * 2);
    float* proj   = (float*)carve((size_t)MROWS * NPROJ * 4);
    float* p_red  = (float*)carve((size_t)MROWS * DMODEL * 4);
    float* gkbuf  = (float*)carve((size_t)MROWS * DKDIM * 4);
    float* khbuf  = (float*)carve((size_t)MROWS * DKDIM * 4);
    float* o_scan = (float*)carve((size_t)BB * HH * LL * HEADV * 4);

    conv_silu_kernel<<<(BB * LL * EMBED) / 256, 256, 0, stream>>>(x, conv_w, xs_h);

    auto cvt = [&](const float* s, _Float16* d, int n) {
        f32_to_f16_kernel<<<(n + 255) / 256, 256, 0, stream>>>(s, d, n);
    };
    cvt(Wq,   Wcat_h + 0,            DKDIM * EMBED);
    cvt(Wkg,  Wcat_h + 256 * EMBED,  DKDIM * EMBED);
    cvt(Wv,   Wcat_h + 512 * EMBED,  EMBED * EMBED);
    cvt(Wg,   Wcat_h + 1024 * EMBED, EMBED * EMBED);
    cvt(Wp,   Wp_h,                  DMODEL * DMODEL);
    cvt(Wout, Wout_h,                EMBED * EMBED);

    reduce_embed_kernel<<<BB * LL, 256, 0, stream>>>(embedqk, ered_h);

    wmma_gemm_kernel<<<dim3(NPROJ / 16, MROWS / 16), 32, 0, stream>>>(
        xs_h, Wcat_h, proj, MROWS, NPROJ, EMBED);

    wmma_gemm_kernel<<<dim3(DMODEL / 16, MROWS / 16), 32, 0, stream>>>(
        ered_h, Wp_h, p_red, MROWS, DMODEL, DMODEL);

    gate_decay_kernel<<<(MROWS * DKDIM) / 256, 256, 0, stream>>>(
        proj, p_red, gkbuf, khbuf);

    gla_chunk_kernel<<<BB * HH, 128, 0, stream>>>(proj, gkbuf, khbuf, o_scan);

    aug_ln_gate_kernel<<<BB * HH * LL, 64, 0, stream>>>(
        proj, khbuf, o_scan, aug, bg, z_h);

    wmma_gemm_kernel<<<dim3(EMBED / 16, MROWS / 16), 32, 0, stream>>>(
        z_h, Wout_h, out, MROWS, EMBED, EMBED);
}
